// StreamingMoELayer_2525440770771
// MI455X (gfx1250) — compile-verified, matrix operand order checked by reading, not run on hardware
//
#include <hip/hip_runtime.h>
#include <hip/hip_bf16.h>
#include <stdint.h>

// ---------------- problem constants (match reference) ----------------
#define NTOK   2048      // BATCH*SEQ tokens
#define DHID   2048      // hidden size
#define FFN    768       // ffn size
#define NEXP   64        // experts
#define TOPK   8
#define NPAIR  (NTOK * TOPK)        // 16384
#define CAP    512                  // 2 * NPAIR / NEXP

// LDS row pitch for bf16 tiles: 40 elems = 80B = 20 dwords.
// 20 mod 64 banks with gcd(20,64)=4 -> 16 consecutive rows hit 16 distinct
// banks during fragment reads. TDM reproduces this pitch in hardware with
// pad_interval=3 (pad after every 16 dwords = one 32-elem bf16 row) and
// pad_amount=3 (insert 4 dwords = 16B).
#define PITCH  40

typedef __attribute__((ext_vector_type(16))) __bf16 v16bf;
typedef __attribute__((ext_vector_type(8)))  float  v8f;
typedef __attribute__((ext_vector_type(4)))  uint32_t v4u;
typedef __attribute__((ext_vector_type(8)))  int      v8i;
typedef __attribute__((ext_vector_type(4)))  int      v4i;

union Frag { uint32_t u[8]; v16bf v; };

__device__ __forceinline__ uint16_t f2bf(float f) {
  uint32_t x = __float_as_uint(f);
  uint32_t r = x + 0x7FFFu + ((x >> 16) & 1u);   // round-to-nearest-even
  return (uint16_t)(r >> 16);
}
__device__ __forceinline__ uint32_t pack2bf(float a, float b) {
#if __has_builtin(__builtin_amdgcn_cvt_pk_bf16_f32)
  auto pk = __builtin_amdgcn_cvt_pk_bf16_f32(a, b);   // v_cvt_pk_bf16_f32
  uint32_t u;
  __builtin_memcpy(&u, &pk, sizeof(u));
  return u;
#else
  return (uint32_t)f2bf(a) | ((uint32_t)f2bf(b) << 16);
#endif
}

// ---------------------------------------------------------------------
// Tensor Data Mover: async-load a [rows x 32] bf16 tile (row stride
// `stride_elems` in the source tensor) into LDS at byte offset `lds_off`,
// with hardware padding producing a 40-bf16 (80B) LDS row pitch.
// D# layout per CDNA5 ISA 8.3/8.4 (group0 128b, group1 256b).
// This toolchain exposes the 6-arg builtin:
//   (v4u g0, v8i g1, v4i g2, v4i g3, v8i extra, i32 cpol)
// groups 2/3 and the extra group are unused for a 2-D tensor -> zeros.
// ---------------------------------------------------------------------
__device__ __forceinline__ void tdm_load_tile_bf16(uint32_t lds_off,
                                                   const uint16_t* gptr,
                                                   uint32_t tensor_k,
                                                   uint32_t rows,
                                                   uint32_t stride_elems) {
  uint64_t ga = (uint64_t)(uintptr_t)gptr;           // 57-bit byte address
  v4u g0;
  g0[0] = 1u;                                        // count=1, user mode
  g0[1] = lds_off;                                   // lds_addr
  g0[2] = (uint32_t)ga;                              // global_addr[31:0]
  g0[3] = (uint32_t)((ga >> 32) & 0x01FFFFFFu)       // global_addr[56:32]
        | (2u << 30);                                // type=2 ("image")
  v8i g1;
  g1[0] = (int)((1u << 16)        // data_size = 1 -> 2-byte elements
              | (1u << 20)        // pad_enable
              | (3u << 22)        // pad_interval: 16 dwords (one 32-elem row)
              | (3u << 25));      // pad_amount: 4 dwords (16B) -> pitch 80B
  g1[1] = (int)((tensor_k & 0xFFFFu) << 16);                 // tensor_dim0 lo16
  g1[2] = (int)(((tensor_k >> 16) & 0xFFFFu)                 // tensor_dim0 hi16
              | ((rows & 0xFFFFu) << 16));                   // tensor_dim1 lo16
  g1[3] = (int)(((rows >> 16) & 0xFFFFu)                     // tensor_dim1 hi16
              | (32u << 16));                                // tile_dim0 = 32
  g1[4] = (int)rows;                                         // tile_dim1; tile_dim2=0
  g1[5] = (int)stride_elems;                                 // tensor_dim0_stride lo32
  g1[6] = 0;                                                 // stride hi; dim1_stride lo
  g1[7] = 0;
  v4i z4 = {0, 0, 0, 0};
  v8i z8 = {0, 0, 0, 0, 0, 0, 0, 0};
  __builtin_amdgcn_tensor_load_to_lds(g0, g1, z4, z4, z8, 0);
}

// ---------------------------------------------------------------------
// 0) zero the output accumulator (harness poisons it; we scatter-add)
// ---------------------------------------------------------------------
__global__ __launch_bounds__(256) void zero_out_kernel(float* out, int n) {
  int i = blockIdx.x * 256 + threadIdx.x;
  if (i < n) out[i] = 0.0f;
}

// ---------------------------------------------------------------------
// 1) gating: logits = x @ gate_w^T, softmax, top-8
//    one 64-thread block (2 waves) per token
// ---------------------------------------------------------------------
__global__ __launch_bounds__(64) void gate_topk_kernel(
    const float* __restrict__ x, const float* __restrict__ gw,
    int* __restrict__ topk_e, float* __restrict__ topk_s) {
  int t   = blockIdx.x;
  int tid = threadIdx.x;
  __shared__ float xs[DHID];
  __shared__ float logits[NEXP];

  for (int d = tid; d < DHID; d += 64) xs[d] = x[(size_t)t * DHID + d];
  __syncthreads();

  const float* w = gw + (size_t)tid * DHID;
  float acc = 0.0f;
  #pragma unroll 4
  for (int d = 0; d < DHID; ++d) acc = fmaf(xs[d], w[d], acc);
  logits[tid] = acc;
  __syncthreads();

  if (tid == 0) {
    float mx = -3.0e38f;
    for (int e = 0; e < NEXP; ++e) mx = fmaxf(mx, logits[e]);
    float sum = 0.0f;
    for (int e = 0; e < NEXP; ++e) {
      float p = __expf(logits[e] - mx);
      logits[e] = p;
      sum += p;
    }
    float inv = __builtin_amdgcn_rcpf(sum);      // v_rcp_f32, no div chain
    for (int k = 0; k < TOPK; ++k) {
      int bi = 0; float bp = -1.0f;
      for (int e = 0; e < NEXP; ++e)
        if (logits[e] > bp) { bp = logits[e]; bi = e; }
      topk_e[t * TOPK + k] = bi;
      topk_s[t * TOPK + k] = bp * inv;
      logits[bi] = -2.0f;
    }
  }
}

// ---------------------------------------------------------------------
// 2) deterministic bucketization: one wave per expert, ballot prefix-sum
//    over all pairs in order -> stable positions (matches stable group-by)
// ---------------------------------------------------------------------
__global__ __launch_bounds__(32) void build_buckets_kernel(
    const int* __restrict__ topk_e, const float* __restrict__ topk_s,
    int* __restrict__ counts, int* __restrict__ btok,
    float* __restrict__ bscore, int* __restrict__ pair_slot) {
  int e    = blockIdx.x;
  int lane = threadIdx.x;
  int offset = 0;
  for (int base = 0; base < NPAIR; base += 32) {
    int p = base + lane;
    bool match = (topk_e[p] == e);
    uint32_t mask = __builtin_amdgcn_ballot_w32(match);
    int pos = offset + __popc(mask & ((1u << lane) - 1u));
    if (match) {
      if (pos < CAP) {
        btok[e * CAP + pos]   = p >> 3;           // source token
        bscore[e * CAP + pos] = topk_s[p];        // routing weight
        pair_slot[p] = e * CAP + pos;
      } else {
        pair_slot[p] = -1;                        // capacity drop
      }
    }
    offset += __popc(mask);
  }
  if (lane == 0) counts[e] = offset < CAP ? offset : CAP;
}

// ---------------------------------------------------------------------
// 3) gather routed rows -> xbuf bf16 [E, CAP, DHID] (paired conversions)
// ---------------------------------------------------------------------
__global__ __launch_bounds__(64) void gather_xbuf_kernel(
    const float* __restrict__ x, const int* __restrict__ pair_slot,
    uint16_t* __restrict__ xbuf) {
  int p = blockIdx.x;
  int slot = pair_slot[p];
  if (slot < 0) return;
  int tok = p >> 3;
  const float* src = x + (size_t)tok * DHID;
  uint16_t*    dst = xbuf + (size_t)slot * DHID;
  for (int d = threadIdx.x * 2; d < DHID; d += 128) {
    float2 f2 = *(const float2*)(src + d);
    *(uint32_t*)(dst + d) = pack2bf(f2.x, f2.y);
  }
}

// ---------------------------------------------------------------------
// WMMA fragment helpers (16-bit A 16x32 / B 32x16 VGPR layout, wave32):
// lane l: row/col = l&15, half = l>>4 ; vgpr v holds K = (v>>2)*16 + half*8
// + (v&3)*2 (+0,+1 packed)
// ---------------------------------------------------------------------
__device__ __forceinline__ void load_frag(const uint16_t* lds, int rc, int half,
                                          Frag& f) {
  #pragma unroll
  for (int v = 0; v < 8; ++v) {
    int k = ((v >> 2) << 4) + (half << 3) + ((v & 3) << 1);
    f.u[v] = *(const uint32_t*)(lds + rc * PITCH + k);
  }
}

// ---------------------------------------------------------------------
// 4) grouped GEMM #1 + SwiGLU:  h = silu(x@Wg^T) * (x@Wu^T)  (bf16 out)
//    block = 256 threads (8 waves), tile = 64 rows x 32 ffn cols,
//    K = DHID in steps of 32; A tile arrives via TDM, B via fp32->bf16
//    VALU conversion; 2 WMMAs (gate,up) per wave per K-step.
// ---------------------------------------------------------------------
__global__ __launch_bounds__(256) void gemm1_swiglu_kernel(
    const uint16_t* __restrict__ xbuf, const float* __restrict__ wg,
    const float* __restrict__ wu, const int* __restrict__ counts,
    uint16_t* __restrict__ hbuf) {
  int e  = blockIdx.z;
  int m0 = blockIdx.y * 64;
  int f0 = blockIdx.x * 32;
  if (m0 >= counts[e]) return;            // uniform per block

  __shared__ __align__(16) uint16_t As[64 * PITCH];
  __shared__ __align__(16) uint16_t Bg[32 * PITCH];
  __shared__ __align__(16) uint16_t Bu[32 * PITCH];

  int tid  = threadIdx.x;
  int lane = tid & 31;
  int wv   = tid >> 5;
  int mt   = wv >> 1;                     // 0..3  (row sub-tile)
  int ft   = wv & 1;                      // 0..1  (col sub-tile)
  int half = lane >> 4;

  uint32_t As_lds = (uint32_t)(uintptr_t)As;
  const uint16_t* arow = xbuf + (size_t)(e * CAP + m0) * DHID;

  v8f accg = {}; v8f accu = {};

  for (int k0 = 0; k0 < DHID; k0 += 32) {
    __syncthreads();                       // prior-iteration readers done
    if (wv == 0) {                         // wave 0 drives the TDM
      tdm_load_tile_bf16(As_lds, arow + k0, DHID, 64, DHID);
    }
    {   // stage Bg/Bu: 32x32 fp32 -> bf16 (4 floats per thread each)
      int fr = tid >> 3;
      int kk = (tid & 7) * 4;
      size_t off = (size_t)(e * FFN + f0 + fr) * DHID + k0 + kk;
      float4 g4 = *(const float4*)(wg + off);
      float4 u4 = *(const float4*)(wu + off);
      if (k0 + 32 < DHID) {               // hint next weight chunk toward L2
        __builtin_prefetch(wg + off + 32, 0, 1);
        __builtin_prefetch(wu + off + 32, 0, 1);
      }
      uint32_t* dg = (uint32_t*)(Bg + fr * PITCH + kk);
      dg[0] = pack2bf(g4.x, g4.y); dg[1] = pack2bf(g4.z, g4.w);
      uint32_t* du = (uint32_t*)(Bu + fr * PITCH + kk);
      du[0] = pack2bf(u4.x, u4.y); du[1] = pack2bf(u4.z, u4.w);
    }
    if (wv == 0) __builtin_amdgcn_s_wait_tensorcnt(0);
    __syncthreads();                       // TDM + B staging visible to all

    Frag a, bg, bu;
    load_frag(As, mt * 16 + (lane & 15), half, a);
    load_frag(Bg, ft * 16 + (lane & 15), half, bg);
    load_frag(Bu, ft * 16 + (lane & 15), half, bu);

    accg = __builtin_amdgcn_wmma_f32_16x16x32_bf16(
        false, a.v, false, bg.v, (short)0, accg, false, false);
    accu = __builtin_amdgcn_wmma_f32_16x16x32_bf16(
        false, a.v, false, bu.v, (short)0, accu, false, false);
  }

  // SwiGLU and store h (bf16).  C/D layout: lanes<16 -> M=i, lanes>=16 -> M=8+i
  int n     = f0 + ft * 16 + (lane & 15);
  int mbase = m0 + mt * 16 + (half << 3);
  #pragma unroll
  for (int i = 0; i < 8; ++i) {
    float g = accg[i], u = accu[i];
    float sig = __builtin_amdgcn_rcpf(1.0f + __expf(-g));  // v_rcp_f32
    hbuf[(size_t)(e * CAP + mbase + i) * FFN + n] = f2bf(g * sig * u);
  }
}

// ---------------------------------------------------------------------
// 5) grouped GEMM #2 + weighted scatter:  out[tok] += s * (h @ Wd^T)
// ---------------------------------------------------------------------
__global__ __launch_bounds__(256) void gemm2_scatter_kernel(
    const uint16_t* __restrict__ hbuf, const float* __restrict__ wd,
    const int* __restrict__ counts, const int* __restrict__ btok,
    const float* __restrict__ bscore, float* __restrict__ out) {
  int e   = blockIdx.z;
  int m0  = blockIdx.y * 64;
  int d0  = blockIdx.x * 32;
  int cnt = counts[e];
  if (m0 >= cnt) return;

  __shared__ __align__(16) uint16_t As[64 * PITCH];
  __shared__ __align__(16) uint16_t Bs[32 * PITCH];

  int tid  = threadIdx.x;
  int lane = tid & 31;
  int wv   = tid >> 5;
  int mt   = wv >> 1;
  int ft   = wv & 1;
  int half = lane >> 4;

  uint32_t As_lds = (uint32_t)(uintptr_t)As;
  const uint16_t* arow = hbuf + (size_t)(e * CAP + m0) * FFN;

  v8f acc = {};

  for (int k0 = 0; k0 < FFN; k0 += 32) {
    __syncthreads();
    if (wv == 0) {                         // async A tile via TDM
      tdm_load_tile_bf16(As_lds, arow + k0, FFN, 64, FFN);
    }
    {   // stage B: 32 cols of Wd, 32 K (fp32 -> bf16)
      int dr = tid >> 3;
      int kk = (tid & 7) * 4;
      size_t off = (size_t)(e * DHID + d0 + dr) * FFN + k0 + kk;
      float4 w4 = *(const float4*)(wd + off);
      if (k0 + 32 < FFN) __builtin_prefetch(wd + off + 32, 0, 1);
      uint32_t* db = (uint32_t*)(Bs + dr * PITCH + kk);
      db[0] = pack2bf(w4.x, w4.y); db[1] = pack2bf(w4.z, w4.w);
    }
    if (wv == 0) __builtin_amdgcn_s_wait_tensorcnt(0);
    __syncthreads();

    Frag a, b;
    load_frag(As, mt * 16 + (lane & 15), half, a);
    load_frag(Bs, ft * 16 + (lane & 15), half, b);
    acc = __builtin_amdgcn_wmma_f32_16x16x32_bf16(
        false, a.v, false, b.v, (short)0, acc, false, false);
  }

  int dcol  = d0 + ft * 16 + (lane & 15);
  int rbase = m0 + mt * 16 + (half << 3);
  #pragma unroll
  for (int i = 0; i < 8; ++i) {
    int r = rbase + i;
    if (r < cnt) {
      int   tok = btok[e * CAP + r];
      float s   = bscore[e * CAP + r];
      atomicAdd(&out[(size_t)tok * DHID + dcol], s * acc[i]);
    }
  }
}

// ---------------------------------------------------------------------
// launch
// ---------------------------------------------------------------------
extern "C" void kernel_launch(void* const* d_in, const int* in_sizes, int n_in,
                              void* d_out, int out_size, void* d_ws, size_t ws_size,
                              hipStream_t stream) {
  const float* x   = (const float*)d_in[0];   // [1,2048,2048]
  const float* gw  = (const float*)d_in[1];   // [64,2048]
  const float* wgt = (const float*)d_in[2];   // [64,768,2048]
  const float* wup = (const float*)d_in[3];   // [64,768,2048]
  const float* wdn = (const float*)d_in[4];   // [64,2048,768]
  float* out = (float*)d_out;                 // [1,2048,2048]

  char* ws = (char*)d_ws;
  size_t o = 0;
  int*      counts    = (int*)(ws + o);      o += 256;
  int*      topk_e    = (int*)(ws + o);      o += (size_t)NPAIR * 4;
  float*    topk_s    = (float*)(ws + o);    o += (size_t)NPAIR * 4;
  int*      pair_slot = (int*)(ws + o);      o += (size_t)NPAIR * 4;
  int*      btok      = (int*)(ws + o);      o += (size_t)NEXP * CAP * 4;
  float*    bscore    = (float*)(ws + o);    o += (size_t)NEXP * CAP * 4;
  o = (o + 255) & ~(size_t)255;
  uint16_t* xbuf      = (uint16_t*)(ws + o); o += (size_t)NEXP * CAP * DHID * 2;
  uint16_t* hbuf      = (uint16_t*)(ws + o); o += (size_t)NEXP * CAP * FFN * 2;

  const int out_n = NTOK * DHID;
  zero_out_kernel<<<(out_n + 255) / 256, 256, 0, stream>>>(out, out_n);
  gate_topk_kernel<<<NTOK, 64, 0, stream>>>(x, gw, topk_e, topk_s);
  build_buckets_kernel<<<NEXP, 32, 0, stream>>>(topk_e, topk_s, counts, btok,
                                                bscore, pair_slot);
  gather_xbuf_kernel<<<NPAIR, 64, 0, stream>>>(x, pair_slot, xbuf);
  gemm1_swiglu_kernel<<<dim3(FFN / 32, CAP / 64, NEXP), 256, 0, stream>>>(
      xbuf, wgt, wup, counts, hbuf);
  gemm2_scatter_kernel<<<dim3(DHID / 32, CAP / 64, NEXP), 256, 0, stream>>>(
      hbuf, wdn, counts, btok, bscore, out);
}
